// SparseMoELayer_60928406061075
// MI455X (gfx1250) — compile-verified, matrix operand order checked by reading, not run on hardware
//
#include <hip/hip_runtime.h>
#include <hip/hip_bf16.h>

// ---------------------------------------------------------------------------
// Sparse MoE forward for MI455X (gfx1250, wave32, WMMA bf16 16x16x32).
// Pipeline: zero -> route(top2) -> seg offsets -> gather/convert ->
//           fc1 grouped GEMM (relu) -> fc2 grouped GEMM (scaled scatter-add).
// Double-buffered LDS pipeline: A-tiles (bf16) via GLOBAL_LOAD_ASYNC_TO_LDS_B128
// (ASYNCcnt), B-tiles (f32 weights) via registers + packed v_cvt_pk_bf16_f32,
// both staged one K-chunk ahead of the WMMA consumer; B prefetched 2 ahead.
// ---------------------------------------------------------------------------

typedef float          v8f     __attribute__((ext_vector_type(8)));
typedef float          f32x4   __attribute__((ext_vector_type(4)));
typedef float          f32x8   __attribute__((ext_vector_type(8)));
typedef __bf16         bf16x4  __attribute__((ext_vector_type(4)));
typedef __bf16         bf16x8  __attribute__((ext_vector_type(8)));
typedef __bf16         bf16x16 __attribute__((ext_vector_type(16)));
typedef unsigned short u16x4   __attribute__((ext_vector_type(4)));
typedef unsigned short u16x8   __attribute__((ext_vector_type(8)));
typedef unsigned short u16x16  __attribute__((ext_vector_type(16)));
typedef int            i32x4   __attribute__((vector_size(16)));   // <4 x i32> for async-LDS builtin

typedef __attribute__((address_space(1))) i32x4* gv4p;   // global 16B vector ptr
typedef __attribute__((address_space(3))) i32x4* lv4p;   // LDS    16B vector ptr

#define D_MODEL 1024
#define HIDDEN  4096
#define NEXP    8
#define NTOK    2048
#define TILE    128
#define KCH     32
#define LDSS    40      // LDS row stride in ushorts: 80B rows -> 16B aligned, conflict-free
#define LDSTILE (TILE * LDSS)
#define MAXROWS 5120    // sum of 128-padded per-expert counts <= 4096 + 8*128

#if __has_builtin(__builtin_amdgcn_global_load_async_to_lds_b128) && \
    __has_builtin(__builtin_amdgcn_s_wait_asynccnt)
#define USE_ASYNC_LDS 1
#else
#define USE_ASYNC_LDS 0
#endif

__device__ __forceinline__ unsigned short bf16_bits(float f) {
  return __builtin_bit_cast(unsigned short, (__bf16)f);   // RNE fptrunc
}

// Stage one 32B slice of the (already bf16) A tile into LDS.
__device__ __forceinline__ void stage_a(const unsigned short* s, unsigned short* d) {
#if USE_ASYNC_LDS
  __builtin_amdgcn_global_load_async_to_lds_b128((gv4p)s,       (lv4p)d,       0, 0);
  __builtin_amdgcn_global_load_async_to_lds_b128((gv4p)(s + 8), (lv4p)(d + 8), 0, 0);
#else
  const u16x8* sv = (const u16x8*)s;
  u16x8 a0 = sv[0], a1 = sv[1];
  *(u16x8*)d       = a0;
  *(u16x8*)(d + 8) = a1;
#endif
}

// Load one 16x32 bf16 fragment (A- or B-side; identical per-lane layout):
// lane<16:  row=lane,    k = {0..7, 16..23}
// lane>=16: row=lane-16, k = {8..15, 24..31}
__device__ __forceinline__ bf16x16 load_frag(const unsigned short* t, int rowBase, int lane) {
  const int r  = lane & 15;
  const int k0 = (lane >> 4) * 8;
  const unsigned short* p = t + (rowBase + r) * LDSS + k0;
  u16x8 lo = *(const u16x8*)p;          // ds_load_b128
  u16x8 hi = *(const u16x8*)(p + 16);   // ds_load_b128
  u16x16 w;
#pragma unroll
  for (int i = 0; i < 8; ++i) { w[i] = lo[i]; w[i + 8] = hi[i]; }
  return __builtin_bit_cast(bf16x16, w);
}

// Shared 128x128 GEMM core: OUT(128x128) = Abf16(128xK) * Bf32(128xK)^T.
// 8 waves in 2x4 layout; each wave: 4x2 frags of 16x16 (f32 accum).
// Double-buffered: chunk c+1 is staged (A: async DMA, B: regs) during chunk c's WMMAs.
template <int KTOT, int ASTR, int BSTR>
__device__ __forceinline__ void gemm_core(const unsigned short* __restrict__ Abase,
                                          const float* __restrict__ Bbase,
                                          unsigned short* As,   // 2 * LDSTILE
                                          unsigned short* Bs,   // 2 * LDSTILE
                                          int tid, v8f acc[4][2]) {
  const int lane = tid & 31, wid = tid >> 5;
  const int wm = wid >> 2, wn = wid & 3;
  const int crow = tid >> 1, coff = (tid & 1) * 16;
  constexpr int NC = KTOT / KCH;

  const unsigned short* aSrc = Abase + (size_t)crow * ASTR + coff;
  const float*          bSrc = Bbase + (size_t)crow * BSTR + coff;
  unsigned short* aD = &As[crow * LDSS + coff];
  unsigned short* bD = &Bs[crow * LDSS + coff];

  // Prologue: stage chunk 0 (A -> LDS buf0 async; B -> regs)
  stage_a(aSrc, aD);
  f32x8 bf0 = ((const f32x8*)bSrc)[0];
  f32x8 bf1 = ((const f32x8*)bSrc)[1];

  for (int c = 0; c < NC; ++c) {
    const int cur = c & 1, nxt = cur ^ 1;
    { // publish B chunk c into LDS[cur] (packed cvt)
      bf16x8 o0 = __builtin_convertvector(bf0, bf16x8);
      bf16x8 o1 = __builtin_convertvector(bf1, bf16x8);
      *(u16x8*)(bD + cur * LDSTILE)     = __builtin_bit_cast(u16x8, o0);
      *(u16x8*)(bD + cur * LDSTILE + 8) = __builtin_bit_cast(u16x8, o1);
    }
#if USE_ASYNC_LDS
    __builtin_amdgcn_s_wait_asynccnt(0);   // A chunk c landed in LDS[cur]
#endif
    __syncthreads();                        // publish; prior readers of [nxt] done
    if (c + 1 < NC) {                       // stage chunk c+1 into [nxt]
      const int k = (c + 1) * KCH;
      stage_a(aSrc + k, aD + nxt * LDSTILE);
      bf0 = ((const f32x8*)(bSrc + k))[0];
      bf1 = ((const f32x8*)(bSrc + k))[1];
      if (c + 2 < NC)
        __builtin_prefetch(bSrc + (c + 2) * KCH, 0, 0);   // weight stream -> L2
    }
    const unsigned short* Ac = As + cur * LDSTILE;
    const unsigned short* Bc = Bs + cur * LDSTILE;
    bf16x16 a[4], b[2];
#pragma unroll
    for (int i = 0; i < 4; ++i) a[i] = load_frag(Ac, wm * 64 + i * 16, lane);
#pragma unroll
    for (int j = 0; j < 2; ++j) b[j] = load_frag(Bc, wn * 32 + j * 16, lane);
#pragma unroll
    for (int i = 0; i < 4; ++i)
#pragma unroll
      for (int j = 0; j < 2; ++j)
        acc[i][j] = __builtin_amdgcn_wmma_f32_16x16x32_bf16(
            false, a[i], false, b[j], (short)0, acc[i][j], false, false);
  }
}

// ---------------------------------------------------------------- zero ------
__global__ __launch_bounds__(256) void zero_kernel(float* __restrict__ out, int n4,
                                                   int* __restrict__ cnt) {
  int i = blockIdx.x * 256 + threadIdx.x;
  if (i < n4) ((float4*)out)[i] = make_float4(0.f, 0.f, 0.f, 0.f);
  if (i < NEXP) cnt[i] = 0;
}

// --------------------------------------------------------------- route ------
// One wave per token: 8 expert logits via lane-parallel dot products,
// renormalized top-2 (softmax ratio == sigmoid of logit difference).
__global__ __launch_bounds__(256) void route_kernel(const float* __restrict__ x,
                                                    const float* __restrict__ gw,
                                                    int* __restrict__ cnt,
                                                    int* __restrict__ tok,
                                                    float* __restrict__ scl) {
  const int wid = threadIdx.x >> 5, lane = threadIdx.x & 31;
  const int t = blockIdx.x * 8 + wid;
  const float4* xr = (const float4*)(x + (size_t)t * D_MODEL);
  const float4* g  = (const float4*)gw;
  float acc[NEXP];
#pragma unroll
  for (int e = 0; e < NEXP; ++e) acc[e] = 0.f;
#pragma unroll
  for (int i = 0; i < 8; ++i) {
    float4 xv = xr[i * 32 + lane];
#pragma unroll
    for (int e = 0; e < NEXP; ++e) {
      float4 gv = g[e * 256 + i * 32 + lane];
      acc[e] += xv.x * gv.x + xv.y * gv.y + xv.z * gv.z + xv.w * gv.w;
    }
  }
#pragma unroll
  for (int e = 0; e < NEXP; ++e)
    for (int off = 16; off; off >>= 1)
      acc[e] += __shfl_xor(acc[e], off, 32);
  if (lane == 0) {
    int e0 = 0;
#pragma unroll
    for (int e = 1; e < NEXP; ++e) if (acc[e] > acc[e0]) e0 = e;
    int e1 = (e0 == 0) ? 1 : 0;
#pragma unroll
    for (int e = 0; e < NEXP; ++e) if (e != e0 && acc[e] > acc[e1]) e1 = e;
    const float s0 = 1.f / (1.f + __expf(acc[e1] - acc[e0]));
    const float s1 = 1.f / (1.f + __expf(acc[e0] - acc[e1]));
    int p0 = atomicAdd(&cnt[e0], 1); tok[e0 * NTOK + p0] = t; scl[e0 * NTOK + p0] = s0;
    int p1 = atomicAdd(&cnt[e1], 1); tok[e1 * NTOK + p1] = t; scl[e1 * NTOK + p1] = s1;
  }
}

// ----------------------------------------------------------------- seg ------
__global__ void seg_kernel(const int* __restrict__ cnt, int* __restrict__ segoff,
                           int* __restrict__ padded) {
  if (threadIdx.x == 0 && blockIdx.x == 0) {
    int off = 0;
    for (int e = 0; e < NEXP; ++e) {
      int pc = ((cnt[e] + TILE - 1) / TILE) * TILE;
      segoff[e] = off; padded[e] = pc; off += pc;
    }
  }
}

// -------------------------------------------------------------- gather ------
// One wave per gathered row: copy x[token] f32 -> bf16 into compact segment;
// zero-fill the 128-row padding so padded GEMM rows are benign.
__global__ __launch_bounds__(256) void gather_kernel(const float* __restrict__ x,
                                                     const int* __restrict__ cnt,
                                                     const int* __restrict__ segoff,
                                                     const int* __restrict__ padded,
                                                     const int* __restrict__ tok,
                                                     unsigned short* __restrict__ Xg) {
  const int e = blockIdx.y;
  const int wid = threadIdx.x >> 5, lane = threadIdx.x & 31;
  const int row = blockIdx.x * 8 + wid;
  if (row >= padded[e]) return;
  unsigned short* dst = Xg + (size_t)(segoff[e] + row) * D_MODEL;
  if (row < cnt[e]) {
    const f32x4* src = (const f32x4*)(x + (size_t)tok[e * NTOK + row] * D_MODEL);
#pragma unroll
    for (int i = 0; i < 8; ++i) {
      f32x4 v = src[i * 32 + lane];
      bf16x4 w = __builtin_convertvector(v, bf16x4);
      *(u16x4*)(dst + (size_t)(i * 32 + lane) * 4) = __builtin_bit_cast(u16x4, w);
    }
  } else {
    u16x4 z = (u16x4)0;
#pragma unroll
    for (int i = 0; i < 8; ++i)
      *(u16x4*)(dst + (size_t)(i * 32 + lane) * 4) = z;
  }
}

// ----------------------------------------------------------------- fc1 ------
// H = relu(Xg[e] @ w1[e]^T + b1[e]) -> bf16.
__global__ __launch_bounds__(256) void fc1_kernel(const unsigned short* __restrict__ Xg,
                                                  const float* __restrict__ w1,
                                                  const float* __restrict__ b1,
                                                  const int* __restrict__ segoff,
                                                  const int* __restrict__ padded,
                                                  unsigned short* __restrict__ Hb) {
  const int e = blockIdx.z, mt = blockIdx.y, nt = blockIdx.x;
  if (mt * TILE >= padded[e]) return;
  const int seg = segoff[e];
  __shared__ unsigned short As[2 * LDSTILE];
  __shared__ unsigned short Bs[2 * LDSTILE];
  const int tid = threadIdx.x, lane = tid & 31, wid = tid >> 5;
  const int wm = wid >> 2, wn = wid & 3;
  v8f acc[4][2] = {};
  gemm_core<D_MODEL, D_MODEL, D_MODEL>(
      Xg + (size_t)(seg + mt * TILE) * D_MODEL,
      w1 + (size_t)e * HIDDEN * D_MODEL + (size_t)(nt * TILE) * D_MODEL,
      As, Bs, tid, acc);

  const int nl = lane & 15, mh = lane >> 4;
#pragma unroll
  for (int j = 0; j < 2; ++j) {
    const int n = nt * TILE + wn * 32 + j * 16 + nl;
    const float bias = b1[e * HIDDEN + n];
#pragma unroll
    for (int i = 0; i < 4; ++i) {
      const int m0 = mt * TILE + wm * 64 + i * 16 + mh * 8;
      size_t base = (size_t)(seg + m0) * HIDDEN + n;
#pragma unroll
      for (int v = 0; v < 8; ++v) {
        float val = acc[i][j][v] + bias;
        val = val > 0.f ? val : 0.f;
        Hb[base + (size_t)v * HIDDEN] = bf16_bits(val);
      }
    }
  }
}

// ----------------------------------------------------------------- fc2 ------
// out[token] += scale * (H[e] @ w2[e]^T + b2[e]) via atomic f32 scatter-add
// (exactly two commutative adds per output element -> deterministic).
__global__ __launch_bounds__(256) void fc2_kernel(const unsigned short* __restrict__ Hb,
                                                  const float* __restrict__ w2,
                                                  const float* __restrict__ b2,
                                                  const int* __restrict__ cnt,
                                                  const int* __restrict__ segoff,
                                                  const int* __restrict__ padded,
                                                  const int* __restrict__ tok,
                                                  const float* __restrict__ scl,
                                                  float* __restrict__ out) {
  const int e = blockIdx.z, mt = blockIdx.y, nt = blockIdx.x;
  if (mt * TILE >= padded[e]) return;
  const int seg = segoff[e], cnt_e = cnt[e];
  __shared__ unsigned short As[2 * LDSTILE];
  __shared__ unsigned short Bs[2 * LDSTILE];
  const int tid = threadIdx.x, lane = tid & 31, wid = tid >> 5;
  const int wm = wid >> 2, wn = wid & 3;
  v8f acc[4][2] = {};
  gemm_core<HIDDEN, HIDDEN, HIDDEN>(
      Hb + (size_t)(seg + mt * TILE) * HIDDEN,
      w2 + (size_t)e * D_MODEL * HIDDEN + (size_t)(nt * TILE) * HIDDEN,
      As, Bs, tid, acc);

  const int nl = lane & 15, mh = lane >> 4;
  float bias[2];
#pragma unroll
  for (int j = 0; j < 2; ++j)
    bias[j] = b2[e * D_MODEL + nt * TILE + wn * 32 + j * 16 + nl];
#pragma unroll
  for (int i = 0; i < 4; ++i) {
    const int m0 = mt * TILE + wm * 64 + i * 16 + mh * 8;
#pragma unroll
    for (int v = 0; v < 8; ++v) {
      const int m = m0 + v;
      if (m < cnt_e) {
        const int   t  = tok[e * NTOK + m];
        const float sc = scl[e * NTOK + m];
        float* orow = out + (size_t)t * D_MODEL;
#pragma unroll
        for (int j = 0; j < 2; ++j) {
          const int n = nt * TILE + wn * 32 + j * 16 + nl;
          atomicAdd(&orow[n], sc * (acc[i][j][v] + bias[j]));
        }
      }
    }
  }
}

// ---------------------------------------------------------------------------
extern "C" void kernel_launch(void* const* d_in, const int* in_sizes, int n_in,
                              void* d_out, int out_size, void* d_ws, size_t ws_size,
                              hipStream_t stream) {
  const float* x    = (const float*)d_in[0];
  const float* gw   = (const float*)d_in[1];
  const float* w1   = (const float*)d_in[2];
  const float* b1   = (const float*)d_in[3];
  const float* w2   = (const float*)d_in[4];
  const float* b2   = (const float*)d_in[5];
  float* out = (float*)d_out;

  char* ws = (char*)d_ws;
  int*            cnt    = (int*)(ws + 0);
  int*            segoff = (int*)(ws + 256);
  int*            padded = (int*)(ws + 512);
  int*            tok    = (int*)(ws + 1024);
  float*          scl    = (float*)(ws + 1024 + NEXP * NTOK * 4);
  unsigned short* Xg     = (unsigned short*)(ws + (256 * 1024));
  unsigned short* Hb     = (unsigned short*)(ws + (size_t)(12 * 1024 * 1024));
  // Xg: 5120*1024*2 = 10 MB ends < 12 MB.  Hb: 5120*4096*2 = 40 MB -> ~52.5 MB total.

  const int n4 = NTOK * D_MODEL / 4;
  zero_kernel  <<<(n4 + 255) / 256, 256, 0, stream>>>(out, n4, cnt);
  route_kernel <<<NTOK / 8, 256, 0, stream>>>(x, gw, cnt, tok, scl);
  seg_kernel   <<<1, 64, 0, stream>>>(cnt, segoff, padded);
  gather_kernel<<<dim3(NTOK / 8, NEXP), 256, 0, stream>>>(x, cnt, segoff, padded, tok, Xg);
  fc1_kernel   <<<dim3(HIDDEN / TILE, NTOK / TILE, NEXP), 256, 0, stream>>>(
                  Xg, w1, b1, segoff, padded, Hb);
  fc2_kernel   <<<dim3(D_MODEL / TILE, NTOK / TILE, NEXP), 256, 0, stream>>>(
                  Hb, w2, b2, cnt, segoff, padded, tok, scl, out);
}
